// IWLayer_5712306504481
// MI455X (gfx1250) — matrix operand order repaired and checked by reading, not compiled, not run on hardware
//
#include <hip/hip_runtime.h>
#include <stdint.h>

#define PDIM 128
#define TN   128
#define LDT  136            // padded bf16 stride: 272B rows, 16B-aligned 8-elem chunks
#define NPTS 131072
#define NUM_DATA_F 131072.0f

typedef __attribute__((ext_vector_type(16))) __bf16 v16bf;
typedef __attribute__((ext_vector_type(8)))  __bf16 v8bf;
typedef __attribute__((ext_vector_type(8)))  float  v8f;

union U16BF { v16bf v; v8bf h[2]; };
union U8BF  { v8bf  v; __bf16 e[8]; };
union U8F   { v8f   v; float  e[8]; };

// ---------------- Tensor Data Mover support (guarded; falls back to manual copy) -----
#if defined(__gfx1250__) && __has_builtin(__builtin_amdgcn_tensor_load_to_lds) && \
    __has_builtin(__builtin_amdgcn_s_wait_tensorcnt)
#define HAVE_TDM 1
#else
#define HAVE_TDM 0
#endif

#if HAVE_TDM
typedef uint32_t u32x4v __attribute__((ext_vector_type(4)));
typedef int32_t  i32x4v __attribute__((ext_vector_type(4)));
typedef int32_t  i32x8v __attribute__((ext_vector_type(8)));

// DMA a 128x128 bf16 row-major matrix from global into LDS, inserting 4 DWORDs of
// padding after every 64 DWORDs (row = 256B) -> LDS rows of LDT=136 bf16 (272B),
// matching the WMMA operand loaders. Issue from ONE wave only; TENSORcnt-tracked.
__device__ __forceinline__ void tdm_stage_matrix(const __bf16* gsrc, __bf16* ldst) {
  uint64_t ga = (uint64_t)reinterpret_cast<uintptr_t>(gsrc);
  uint32_t la = (uint32_t)reinterpret_cast<uintptr_t>(ldst);   // addr[31:0] = LDS offset
  // D# group0: count=1 | lds_addr | global_addr[56:0] | type=2
  u32x4v g0 = { 1u,
                la,
                (uint32_t)ga,
                ((uint32_t)(ga >> 32) & 0x01FFFFFFu) | 0x80000000u };
  // D# group1: data_size=2B(enc 1), pad_enable, pad_interval=64dw(enc 5),
  //            pad_amount=4dw(enc 3); tensor_dim0=128, tensor_dim1=128,
  //            tile_dim0=128, tile_dim1=128, tensor_dim0_stride=128
  const uint32_t w0 = (1u << 16) | (1u << 20) | (5u << 22) | (3u << 25);
  i32x8v g1 = { (int)w0,
                (int)(128u << 16),   // [47:32]=0, tensor_dim0[15:0]=128
                (int)(128u << 16),   // tensor_dim0[31:16]=0, tensor_dim1[15:0]=128
                (int)(128u << 16),   // tensor_dim1[31:16]=0, tile_dim0=128
                (int)128,            // tile_dim1=128, tile_dim2=0
                (int)128,            // tensor_dim0_stride[31:0]=128
                0, 0 };
  i32x4v z4 = { 0, 0, 0, 0 };
#if __clang_major__ >= 23
  i32x8v z8 = { 0, 0, 0, 0, 0, 0, 0, 0 };
  __builtin_amdgcn_tensor_load_to_lds(g0, g1, z4, z4, z8, 0);
#else
  __builtin_amdgcn_tensor_load_to_lds(g0, g1, z4, z4, 0);
#endif
}
#endif // HAVE_TDM

// ---------------- counter-based RNG (deterministic, stateless) ----------------
__device__ __forceinline__ uint32_t hash3(uint32_t a, uint32_t b, uint32_t c) {
  uint32_t h = (a * 0x9E3779B1u) ^ (b * 0x85EBCA77u) ^ (c * 0xC2B2AE3Du);
  h ^= h >> 16; h *= 0x7FEB352Du;
  h ^= h >> 15; h *= 0x846CA68Bu;
  h ^= h >> 16;
  return h;
}
__device__ __forceinline__ float u01(uint32_t h) {
  return ((float)(h >> 8) + 0.5f) * (1.0f / 16777216.0f);
}
__device__ __forceinline__ float rng_normal(uint32_t a, uint32_t b, uint32_t seed) {
  float u1 = u01(hash3(a, b, seed));
  float u2 = u01(hash3(a, b, seed ^ 0xB5297A4Du));
  return sqrtf(-2.0f * __logf(u1)) * __cosf(6.28318530718f * u2);
}
// Marsaglia-Tsang Gamma(alpha,1), alpha > 1 here; bounded retries, mean fallback
__device__ __forceinline__ float rng_gamma(float alpha, uint32_t a, uint32_t b, uint32_t seed) {
  float d = alpha - (1.0f / 3.0f);
  float c = rsqrtf(9.0f * d);
  for (int it = 0; it < 8; ++it) {
    float x = rng_normal(a, b, seed + 101u * (uint32_t)it);
    float v = 1.0f + c * x;
    if (v <= 0.0f) continue;
    v = v * v * v;
    float u = u01(hash3(a ^ 0x632BE5ABu, b, seed + 131u * (uint32_t)it));
    if (__logf(u) < 0.5f * x * x + d - d * v + d * __logf(v)) return d * v;
  }
  return alpha;
}

// ---------------- WMMA operand loaders (bf16, per ISA 7.12.2 layouts) ----------------
// A 16x32: lane sub = row; halves hi=0/1 pick K {0..7,16..23} / {8..15,24..31}
__device__ __forceinline__ v16bf load_amat(const __bf16* s, int m0, int kk, int sub, int hi) {
  U16BF r;
  const __bf16* base = s + (m0 + sub) * LDT + kk + hi * 8;
  r.h[0] = *reinterpret_cast<const v8bf*>(base);
  r.h[1] = *reinterpret_cast<const v8bf*>(base + 16);
  return r.v;
}
// B 32x16 from transposed-[n][k] LDS tile: lane sub = col; halves hi=0/1 pick K {0..7,16..23} / {8..15,24..31}
__device__ __forceinline__ v16bf load_bmatT(const __bf16* s, int n0, int kk, int sub, int hi) {
  U16BF r;
  const __bf16* base = s + (n0 + sub) * LDT + kk + hi * 8;
  r.h[0] = *reinterpret_cast<const v8bf*>(base);
  r.h[1] = *reinterpret_cast<const v8bf*>(base + 16);
  return r.v;
}
// Out[m0:m0+16, 0:128] = A[m0:m0+16, :] @ X  (X given transposed [n][k] in LDS)
__device__ __forceinline__ void gemm_rowband(const __bf16* sA, const __bf16* sXT,
                                             int m0, int sub, int hi, U8F acc[8]) {
#pragma unroll
  for (int t = 0; t < 8; ++t)
#pragma unroll
    for (int r = 0; r < 8; ++r) acc[t].e[r] = 0.0f;
#pragma unroll
  for (int kk = 0; kk < PDIM; kk += 32) {
    v16bf amat = load_amat(sA, m0, kk, sub, hi);
#pragma unroll
    for (int t = 0; t < 8; ++t) {
      v16bf bmat = load_bmatT(sXT, t * 16, kk, sub, hi);
      acc[t].v = __builtin_amdgcn_wmma_f32_16x16x32_bf16(
          false, amat, false, bmat, (short)0, acc[t].v, false, false);
    }
  }
}

// ---------------- small dense helpers (128 threads, LDS matrices stride 128) ----------------
__device__ void chol128(float* A, int tid) {
  for (int k = 0; k < PDIM; ++k) {
    if (tid == k) A[k * PDIM + k] = sqrtf(fmaxf(A[k * PDIM + k], 1e-20f));
    __syncthreads();
    if (tid > k) A[tid * PDIM + k] /= A[k * PDIM + k];
    __syncthreads();
    if (tid > k) {
      float lik = A[tid * PDIM + k];
      for (int j = k + 1; j <= tid; ++j) A[tid * PDIM + j] -= lik * A[j * PDIM + k];
    }
    __syncthreads();
  }
  for (int j = tid + 1; j < PDIM; ++j) A[tid * PDIM + j] = 0.0f;
  __syncthreads();
}
// lower-triangular inverse, thread tid owns column tid
__device__ void trinv128(const float* L, float* I, int tid) {
  int c = tid;
  for (int r = 0; r < c; ++r) I[r * PDIM + c] = 0.0f;
  I[c * PDIM + c] = 1.0f / L[c * PDIM + c];
  for (int r = c + 1; r < PDIM; ++r) {
    float s = 0.0f;
    for (int j = c; j < r; ++j) s += L[r * PDIM + j] * I[j * PDIM + c];
    I[r * PDIM + c] = -s / L[r * PDIM + r];
  }
  __syncthreads();
}
__device__ float block_sum128(float v, float* scratch, int tid) {
  if (tid == 0) scratch[0] = 0.0f;
  __syncthreads();
  atomicAdd(&scratch[0], v);
  __syncthreads();
  float r = scratch[0];
  __syncthreads();
  return r;
}

// =====================================================================================
// Stage A: all P x P work. One block, 128 threads. Exports d_ii, Kii_inv, Linv^T, Gii
// (bf16) to workspace; writes Gii (f32) and loss to d_out.
// =====================================================================================
extern "C" __global__ void __launch_bounds__(128)
iw_small_kernel(const float* __restrict__ Kii, const float* __restrict__ Vtri,
                const float* __restrict__ p_delta, const float* __restrict__ p_diagp,
                const float* __restrict__ p_gammap, const float* __restrict__ p_height,
                float* __restrict__ outGii, float* __restrict__ outLoss,
                float* __restrict__ ws_dii, __bf16* __restrict__ wsM1,
                __bf16* __restrict__ wsM2, __bf16* __restrict__ wsM3)
{
  extern __shared__ char smem_raw[];
  float* sP   = reinterpret_cast<float*>(smem_raw); // prior_psi (pristine)
  float* sW   = sP + PDIM * PDIM;                   // working / chol target
  float* sI   = sW + PDIM * PDIM;                   // inverses / cache
  float* sM   = sI + PDIM * PDIM;                   // temps
  float* sDii = sM + PDIM * PDIM;                   // [P]
  float* sRed = sDii + PDIM;                        // [4]

  const int tid = threadIdx.x;
  const float delta  = p_delta[0];
  const float diagp  = p_diagp[0];
  const float gammap = p_gammap[0];
  const float height = p_height[0];
  const float prior_nu = delta + (float)PDIM + 1.0f;
  const float post_nu  = prior_nu + gammap;

  // 1) d_ii, prior_psi = delta * (height*exp(-0.5 d2) + jitter I)
  float dii_t = Kii[tid * PDIM + tid];
  sDii[tid] = dii_t;
  ws_dii[tid] = dii_t;
  __syncthreads();
  for (int j = 0; j < PDIM; ++j) {
    float kv = height * __expf(-0.5f * (dii_t + sDii[j] - 2.0f * Kii[tid * PDIM + j]));
    if (j == tid) kv += 1.0e-6f;
    sP[tid * PDIM + j] = delta * kv;
  }
  __syncthreads();

  // 2) L = chol(prior_psi)
  for (int j = 0; j < PDIM; ++j) sW[tid * PDIM + j] = sP[tid * PDIM + j];
  __syncthreads();
  chol128(sW, tid);
  float logdet_prior = 2.0f * block_sum128(__logf(sW[tid * PDIM + tid]), sRed, tid);

  // 3) Linv; export M2 = Linv^T
  trinv128(sW, sI, tid);
  for (int j = 0; j < PDIM; ++j) wsM2[tid * PDIM + j] = (__bf16)sI[j * PDIM + tid];
  // 4) Kii_inv = Linv^T Linv; export M1
  for (int j = 0; j < PDIM; ++j) {
    int k0 = (tid > j) ? tid : j;
    float s = 0.0f;
    for (int k = k0; k < PDIM; ++k) s += sI[k * PDIM + tid] * sI[k * PDIM + j];
    wsM1[tid * PDIM + j] = (__bf16)s;
  }
  __syncthreads();

  // 5) post_psi = prior_psi + (diagp/P) V V^T  (cache V rows in sI)
  for (int j = 0; j < PDIM; ++j) sI[tid * PDIM + j] = Vtri[tid * PDIM + j];
  __syncthreads();
  const float vs = diagp / (float)PDIM;
  for (int j = 0; j < PDIM; ++j) {
    float s = 0.0f;
    for (int k = 0; k < PDIM; ++k) s += sI[tid * PDIM + k] * sI[j * PDIM + k];
    sW[tid * PDIM + j] = sP[tid * PDIM + j] + vs * s;
  }
  __syncthreads();

  // 6) Lq = chol(post_psi)
  chol128(sW, tid);
  float logdet_post = 2.0f * block_sum128(__logf(sW[tid * PDIM + tid]), sRed, tid);

  // 7) Bartlett factor A in sM; tr(A A^T)
  float rowsq = 0.0f;
  for (int j = 0; j < PDIM; ++j) {
    float a;
    if (j < tid)       a = rng_normal((uint32_t)tid, (uint32_t)j, 0xC0FFEE01u);
    else if (j == tid) a = sqrtf(2.0f * rng_gamma(0.5f * (post_nu - (float)tid),
                                                  (uint32_t)tid, 7777u, 0xC0FFEE02u));
    else               a = 0.0f;
    sM[tid * PDIM + j] = a;
    rowsq += a * a;
  }
  __syncthreads();
  float trace_AAT = block_sum128(rowsq, sRed, tid);

  // 8) Ainv
  trinv128(sM, sI, tid);
  // 9) Mfac = Lq @ Ainv^T  (both lower)
  for (int j = 0; j < PDIM; ++j) {
    int kmax = (tid < j) ? tid : j;
    float s = 0.0f;
    for (int k = 0; k <= kmax; ++k) s += sW[tid * PDIM + k] * sI[j * PDIM + k];
    sM[tid * PDIM + j] = s;
  }
  __syncthreads();
  // 10) Gii = Mfac Mfac^T; export
  for (int j = 0; j < PDIM; ++j) {
    float s = 0.0f;
    for (int k = 0; k < PDIM; ++k) s += sM[tid * PDIM + k] * sM[j * PDIM + k];
    sI[tid * PDIM + j] = s;
  }
  __syncthreads();
  for (int j = 0; j < PDIM; ++j) {
    float g = sI[tid * PDIM + j];
    outGii[tid * PDIM + j] = g;
    wsM3[tid * PDIM + j] = (__bf16)g;
  }
  __syncthreads();

  // 11) Lx = chol(Gii)
  chol128(sI, tid);
  float logdet_G = 2.0f * block_sum128(__logf(sI[tid * PDIM + tid]), sRed, tid);
  // 12) Lxinv
  trinv128(sI, sM, tid);
  // 13) tr(Gii^-1 prior_psi) = sum_k (Lxinv S Lxinv^T)[k][k]
  float tloc = 0.0f;
  for (int i = 0; i <= tid; ++i) {
    float li = sM[tid * PDIM + i];
    float inner = 0.0f;
    for (int j = 0; j <= tid; ++j) inner += sP[i * PDIM + j] * sM[tid * PDIM + j];
    tloc += li * inner;
  }
  float traceP = block_sum128(tloc, sRed, tid);

  // 14) multivariate log-gamma sums
  float lg_post  = block_sum128(lgammaf(0.5f * post_nu  - 0.5f * (float)tid), sRed, tid);
  float lg_prior = block_sum128(lgammaf(0.5f * prior_nu - 0.5f * (float)tid), sRed, tid);

  if (tid == 0) {
    const float LOG2  = 0.69314718056f;
    const float LOGPI = 1.14472988585f;
    float mvl_const = LOGPI * (float)PDIM * (float)(PDIM - 1) * 0.25f;
    float logQ = -0.5f * (post_nu + PDIM + 1.0f) * logdet_G
               + 0.5f * post_nu * logdet_post
               - 0.5f * trace_AAT
               - 0.5f * post_nu * (float)PDIM * LOG2
               - (mvl_const + lg_post);
    float logP = -0.5f * (prior_nu + PDIM + 1.0f) * logdet_G
               + 0.5f * prior_nu * logdet_prior
               - 0.5f * traceP
               - 0.5f * prior_nu * (float)PDIM * LOG2
               - (mvl_const + lg_prior);
    outLoss[0] = -(logP - logQ) / NUM_DATA_F;
  }
}

// =====================================================================================
// Stage B: fused N-path. One block per 128-column tile; 8 waves; 3 bf16 WMMA GEMMs.
// Small matrices staged global->LDS via the Tensor Data Mover (TENSORcnt-tracked,
// DMA applies the padded-row LDS layout); falls back to manual copies if no TDM.
// =====================================================================================
extern "C" __global__ void __launch_bounds__(256)
iw_big_kernel(const float* __restrict__ Kit, const float* __restrict__ Ktt,
              const float* __restrict__ p_delta, const float* __restrict__ p_height,
              const float* __restrict__ dii, const __bf16* __restrict__ M1,
              const __bf16* __restrict__ M2, const __bf16* __restrict__ M3,
              float* __restrict__ outGit, float* __restrict__ outGtt)
{
  extern __shared__ char smem_raw[];
  __bf16* sA = reinterpret_cast<__bf16*>(smem_raw);  // [P][LDT]  A-role small matrix
  __bf16* sB = sA + PDIM * LDT;                      // [TN][LDT] X^T tile (B-role)
  __bf16* sC = sB + TN * LDT;                        // [TN][LDT] intermediate^T (B-role)
  float* sDii = reinterpret_cast<float*>(sC + TN * LDT);
  float* sKtt = sDii + PDIM;
  float* sDot = sKtt + TN;
  float* sG   = sDot + TN;
  float* sGtt = sG + TN;

  const int tid  = threadIdx.x;
  const int wave = tid >> 5;
  const int lane = tid & 31;
  const int sub  = lane & 15;
  const int hi   = lane >> 4;
  const int col0 = blockIdx.x * TN;
  const int m0   = wave * 16;

  const float delta  = p_delta[0];
  const float height = p_height[0];
  const float dh = delta * height;   // == dktt scalar (diag of sq-exp kernel is height)

  // warm L2/WGP$ for the shared small matrices
  __builtin_prefetch(M2 + (tid << 3), 0, 1);
  __builtin_prefetch(M3 + (tid << 3), 0, 1);

  // ---- phase 0a: vectors, zero sDot, M1 -> sA (TDM wave-0 DMA or manual) ----
#if HAVE_TDM
  if (tid < 32) tdm_stage_matrix(M1, sA);
#else
  for (int idx = tid; idx < PDIM * 16; idx += 256) {
    int r = idx >> 4, c8 = (idx & 15) << 3;
    *reinterpret_cast<v8bf*>(sA + r * LDT + c8) =
        *reinterpret_cast<const v8bf*>(M1 + r * PDIM + c8);
  }
#endif
  if (tid < PDIM) sDii[tid] = dii[tid];
  if (tid < TN)  { sKtt[tid] = Ktt[col0 + tid]; sDot[tid] = 0.0f; }
  __syncthreads();

  // ---- phase 0b: Kit tile -> dKit^T (bf16) in sB (overlaps the TDM DMA) ----
  for (int idx = tid; idx < PDIM * (TN / 4); idx += 256) {
    int p  = idx >> 5;
    int n4 = (idx & 31) << 2;
    float4 k4 = *reinterpret_cast<const float4*>(Kit + (size_t)p * NPTS + col0 + n4);
    float ke[4] = {k4.x, k4.y, k4.z, k4.w};
    float di = sDii[p];
#pragma unroll
    for (int e = 0; e < 4; ++e) {
      int n = n4 + e;
      float v = dh * __expf(-0.5f * (di + sKtt[n] - 2.0f * ke[e]));
      sB[n * LDT + p] = (__bf16)v;
    }
  }
#if HAVE_TDM
  if (tid < 32) __builtin_amdgcn_s_wait_tensorcnt(0);
#endif
  __syncthreads();

  // ---- GEMM1: T1 = Kii_inv @ dKit; fuse Schur-diagonal dot ----
  U8F acc[8];
  gemm_rowband(sA, sB, m0, sub, hi, acc);
#pragma unroll
  for (int t = 0; t < 8; ++t) {
    int n = t * 16 + sub;
    U8BF dk; dk.v = *reinterpret_cast<const v8bf*>(sB + n * LDT + m0 + hi * 8);
    U8BF pk;
    float part = 0.0f;
#pragma unroll
    for (int r = 0; r < 8; ++r) {
      part += acc[t].e[r] * (float)dk.e[r];
      pk.e[r] = (__bf16)acc[t].e[r];
    }
    *reinterpret_cast<v8bf*>(sC + n * LDT + m0 + hi * 8) = pk.v;  // T1^T
    atomicAdd(&sDot[n], part);
  }
  __syncthreads();

  // ---- phase 2: M2 -> sA (TDM); gtti per column; noise^T -> sB ----
#if HAVE_TDM
  if (tid < 32) tdm_stage_matrix(M2, sA);
#else
  for (int idx = tid; idx < PDIM * 16; idx += 256) {
    int r = idx >> 4, c8 = (idx & 15) << 3;
    *reinterpret_cast<v8bf*>(sA + r * LDT + c8) =
        *reinterpret_cast<const v8bf*>(M2 + r * PDIM + c8);
  }
#endif
  if (tid < TN) {
    float dktti = dh - sDot[tid];
    float alphaN = 0.5f * (delta + (float)PDIM + (float)NPTS + 1.0f);
    float gs = rng_gamma(alphaN, (uint32_t)(col0 + tid), 0xABCDu, 0x00700777u);
    float gtti = 0.5f * dktti / gs;
    sGtt[tid] = gtti;                       // Gtt accumulator starts at gtti
    sG[tid] = sqrtf(fmaxf(gtti, 0.0f));
  }
  for (int idx = tid; idx < PDIM * PDIM; idx += 256) {
    int n = idx >> 7, p = idx & 127;
    sB[n * LDT + p] = (__bf16)rng_normal((uint32_t)p, (uint32_t)(col0 + n), 0x51ED270Bu);
  }
#if HAVE_TDM
  if (tid < 32) __builtin_amdgcn_s_wait_tensorcnt(0);
#endif
  __syncthreads();

  // ---- GEMM2: w = Linv^T @ noise; fuse inv_Gii_git = T1 + w*sqrt(gtti) into sC ----
  gemm_rowband(sA, sB, m0, sub, hi, acc);
#pragma unroll
  for (int t = 0; t < 8; ++t) {
    int n = t * 16 + sub;
    float g = sG[n];
    U8BF cu; cu.v = *reinterpret_cast<const v8bf*>(sC + n * LDT + m0 + hi * 8);
#pragma unroll
    for (int r = 0; r < 8; ++r)
      cu.e[r] = (__bf16)((float)cu.e[r] + acc[t].e[r] * g);
    *reinterpret_cast<v8bf*>(sC + n * LDT + m0 + hi * 8) = cu.v;
  }
  __syncthreads();

  // ---- M3 (Gii) -> sA ----
#if HAVE_TDM
  if (tid < 32) {
    tdm_stage_matrix(M3, sA);
    __builtin_amdgcn_s_wait_tensorcnt(0);
  }
#else
  for (int idx = tid; idx < PDIM * 16; idx += 256) {
    int r = idx >> 4, c8 = (idx & 15) << 3;
    *reinterpret_cast<v8bf*>(sA + r * LDT + c8) =
        *reinterpret_cast<const v8bf*>(M3 + r * PDIM + c8);
  }
#endif
  __syncthreads();

  // ---- GEMM3: Git = Gii @ inv_Gii_git; fuse Gtt column dot ----
  gemm_rowband(sA, sC, m0, sub, hi, acc);
#pragma unroll
  for (int t = 0; t < 8; ++t) {
    int n = t * 16 + sub;
    U8BF cu; cu.v = *reinterpret_cast<const v8bf*>(sC + n * LDT + m0 + hi * 8);
    float part = 0.0f;
#pragma unroll
    for (int r = 0; r < 8; ++r) {
      int row = m0 + hi * 8 + r;
      outGit[(size_t)row * NPTS + col0 + n] = acc[t].e[r];
      part += acc[t].e[r] * (float)cu.e[r];
    }
    atomicAdd(&sGtt[n], part);
  }
  __syncthreads();
  if (tid < TN) outGtt[col0 + tid] = sGtt[tid];
}

// =====================================================================================
extern "C" void kernel_launch(void* const* d_in, const int* in_sizes, int n_in,
                              void* d_out, int out_size, void* d_ws, size_t ws_size,
                              hipStream_t stream) {
  (void)in_sizes; (void)n_in; (void)out_size; (void)ws_size;
  const float* Kii    = (const float*)d_in[0];
  const float* Kit    = (const float*)d_in[1];
  const float* Ktt    = (const float*)d_in[2];
  const float* Vtri   = (const float*)d_in[3];
  const float* delta  = (const float*)d_in[4];
  const float* diagp  = (const float*)d_in[5];
  const float* gammap = (const float*)d_in[6];
  const float* height = (const float*)d_in[7];

  float* out     = (float*)d_out;
  float* outGii  = out;
  float* outGit  = out + PDIM * PDIM;
  float* outGtt  = out + PDIM * PDIM + (size_t)PDIM * NPTS;
  float* outLoss = outGtt + NPTS;

  char* ws = (char*)d_ws;
  float*  ws_dii = (float*)ws;
  __bf16* wsM1   = (__bf16*)(ws + 512);
  __bf16* wsM2   = (__bf16*)(ws + 512 + 32768);
  __bf16* wsM3   = (__bf16*)(ws + 512 + 65536);

  size_t smallLds = (size_t)(4 * PDIM * PDIM + PDIM + 8) * sizeof(float);
  iw_small_kernel<<<1, 128, smallLds, stream>>>(Kii, Vtri, delta, diagp, gammap, height,
                                                outGii, outLoss, ws_dii, wsM1, wsM2, wsM3);

  size_t bigLds = (size_t)3 * PDIM * LDT * sizeof(__bf16)
                + (size_t)(PDIM + 4 * TN) * sizeof(float);
  iw_big_kernel<<<NPTS / TN, 256, bigLds, stream>>>(Kit, Ktt, delta, height,
                                                    ws_dii, wsM1, wsM2, wsM3,
                                                    outGit, outGtt);
}
